// VQ_VAE_20830591386205
// MI455X (gfx1250) — matrix operand order repaired and checked by reading, not compiled, not run on hardware
//
#include <hip/hip_runtime.h>
#include <math.h>

#ifndef __has_builtin
#define __has_builtin(x) 0
#endif
#if __has_builtin(__builtin_amdgcn_tensor_load_to_lds)
#define HAVE_TDM 1
#else
#define HAVE_TDM 0
#endif

// ---------------- types ----------------
typedef __attribute__((ext_vector_type(8)))  unsigned short v8u16;
typedef __attribute__((ext_vector_type(16))) unsigned short v16u16;
typedef __attribute__((ext_vector_type(16))) __bf16        v16bf;
typedef __attribute__((ext_vector_type(8)))  float         v8f;
typedef __attribute__((ext_vector_type(4)))  unsigned int  u32x4;
typedef __attribute__((ext_vector_type(8)))  int           i32x8;
typedef __attribute__((ext_vector_type(4)))  int           i32x4;

constexpr int BM   = 64;   // block tile M
constexpr int BN   = 64;   // block tile N
constexpr int BKT  = 32;   // K step (one bf16 WMMA)
constexpr int LSTR = 40;   // LDS row stride (bf16) -> 80B rows (64B data + 16B pad)

__device__ __forceinline__ unsigned short f2bfs(float f) {  // RNE f32->bf16
  unsigned int u = __float_as_uint(f);
  return (unsigned short)((u + 0x7FFFu + ((u >> 16) & 1u)) >> 16);
}
__device__ __forceinline__ unsigned int pack2bf(float a, float b) {
  unsigned int ua = __float_as_uint(a), ub = __float_as_uint(b);
  ua = (ua + 0x7FFFu + ((ua >> 16) & 1u)) >> 16;
  ub = (ub + 0x7FFFu + ((ub >> 16) & 1u)) & 0xFFFF0000u;
  return ua | ub;
}

// C = act( alpha * A(MxK) @ op(B) + bias + resid )
// BMODE 0: B = f32 (N x K) row-major (NT, activations e.g. K^T)
// BMODE 1: B = f32 (K x N) row-major (NN, P@V; transposed while staging)
// BMODE 2: B = bf16 (N x K) row-major (weights); tile fetched by the Tensor Data Mover
// REQUIRES: M%64==0, N%64==0, K%32==0, rows 16B-aligned (host pads everything).
template <int BMODE, bool BGELU, bool BRES>
__global__ void __launch_bounds__(256)
gemm_wmma(const float* __restrict__ A, const void* __restrict__ Bmv,
          const float* __restrict__ bias, const float* __restrict__ resid,
          float* __restrict__ C, int M, int N, int K,
          int lda, int ldb, int ldc,
          long long aO, long long aI, long long bO, long long bI,
          long long cO, long long cI, int inner, float alpha) {
  __shared__ unsigned short As[BM * LSTR];
  __shared__ unsigned short Bs[BN * LSTR];

  const int z = blockIdx.z;
  const long long zo = (long long)(z / inner), zi = (long long)(z % inner);
  A += zo * aO + zi * aI;
  const float*          Bf = (BMODE != 2) ? (const float*)Bmv + zo * bO + zi * bI : nullptr;
  const unsigned short* Bh = (BMODE == 2) ? (const unsigned short*)Bmv + zo * bO + zi * bI : nullptr;
  const long long coff = zo * cO + zi * cI;
  C += coff;
  const float* Rp = BRES ? resid + coff : nullptr;

  const int tid = threadIdx.x;
  const int gm  = blockIdx.y * BM, gn = blockIdx.x * BN;
  const int wid = tid >> 5, lane = tid & 31;
  const int mw  = wid >> 1, nw = wid & 1;   // 4x2 waves; each wave owns 16x32 of C
  const int l15 = lane & 15, lh = lane >> 4;

  const int ar = tid >> 2, ac = (tid & 3) * 8;    // A / B-row-major loader (64 x 32)
  const int tn = tid & 63, tkg = (tid >> 6) * 8;  // B-NN loader coords

  v8f acc0 = {}, acc1 = {};

  for (int k0 = 0; k0 < K; k0 += BKT) {
    if (BMODE == 2) {
#if HAVE_TDM
      if (wid == 0) {  // one TDM descriptor fetches the whole 64x32 bf16 tile
        const unsigned short* bw = Bh + (long long)gn * ldb + k0;
        unsigned long long ga = (unsigned long long)bw;
        u32x4 g0;
        g0[0] = 1u;                                          // count=1 (valid user D#)
        g0[1] = (unsigned int)(unsigned long long)&Bs[0];    // lds_addr
        g0[2] = (unsigned int)(ga & 0xFFFFFFFFu);            // global_addr[31:0]
        g0[3] = (unsigned int)((ga >> 32) & 0x01FFFFFFu) | (2u << 30);  // addr[56:32]|type=2
        i32x8 g1;
        g1[0] = (int)((1u << 16) | (1u << 20) | (3u << 22) | (3u << 25));
        //        data_size=2B    pad_enable   every 16 DW    pad 4 DW (64B rows -> 80B)
        g1[1] = (int)(32u << 16);   // tensor_dim0 = 32 (K elems per row)
        g1[2] = (int)(64u << 16);   // tensor_dim1 = 64 rows
        g1[3] = (int)(32u << 16);   // tile_dim0 = 32
        g1[4] = 64;                 // tile_dim1 = 64
        g1[5] = ldb;                // tensor_dim0_stride (elements)
        g1[6] = 0;
        g1[7] = 0;
        i32x4 z4 = {0, 0, 0, 0};
#if __clang_major__ >= 23
        i32x8 z8 = {0, 0, 0, 0, 0, 0, 0, 0};
        __builtin_amdgcn_tensor_load_to_lds(g0, g1, z4, z4, z8, 0);
#else
        __builtin_amdgcn_tensor_load_to_lds(g0, g1, z4, z4, 0);
#endif
        __builtin_amdgcn_s_wait_tensorcnt((short)0);
      }
#else
      {  // fallback: direct bf16 vector loads
        const unsigned short* s = Bh + (long long)(gn + ar) * ldb + (k0 + ac);
        *(uint4*)&Bs[ar * LSTR + ac] = *(const uint4*)s;
      }
#endif
    } else if (BMODE == 0) {  // f32 NT tile
      const float* s = Bf + (long long)(gn + ar) * ldb + (k0 + ac);
      float4 f0 = ((const float4*)s)[0];
      float4 f1 = ((const float4*)s)[1];
      uint4 p;
      p.x = pack2bf(f0.x, f0.y); p.y = pack2bf(f0.z, f0.w);
      p.z = pack2bf(f1.x, f1.y); p.w = pack2bf(f1.z, f1.w);
      *(uint4*)&Bs[ar * LSTR + ac] = p;
    } else {  // f32 NN tile -> transpose while staging
      const float* s = Bf + (long long)(k0 + tkg) * ldb + (gn + tn);
      float f[8];
#pragma unroll
      for (int j = 0; j < 8; ++j) f[j] = s[(long long)j * ldb];
      uint4 p;
      p.x = pack2bf(f[0], f[1]); p.y = pack2bf(f[2], f[3]);
      p.z = pack2bf(f[4], f[5]); p.w = pack2bf(f[6], f[7]);
      *(uint4*)&Bs[tn * LSTR + tkg] = p;
    }
    {  // A tile 64x32 f32 -> bf16
      const float* s = A + (long long)(gm + ar) * lda + (k0 + ac);
      float4 f0 = ((const float4*)s)[0];
      float4 f1 = ((const float4*)s)[1];
      uint4 p;
      p.x = pack2bf(f0.x, f0.y); p.y = pack2bf(f0.z, f0.w);
      p.z = pack2bf(f1.x, f1.y); p.w = pack2bf(f1.z, f1.w);
      *(uint4*)&As[ar * LSTR + ac] = p;
    }
    __syncthreads();

    // A frag (ISA 16-bit 16x32 layout): K = lh*8+0..7, then 16+lh*8+0..7
    const unsigned short* ap = &As[(mw * 16 + l15) * LSTR + lh * 8];
    v8u16 a0 = *(const v8u16*)ap;
    v8u16 a1 = *(const v8u16*)(ap + 16);
    v16u16 a16 = __builtin_shufflevector(a0, a1, 0,1,2,3,4,5,6,7,8,9,10,11,12,13,14,15);
    v16bf af = __builtin_bit_cast(v16bf, a16);
    // B frags: K = lh*8+0..7, then 16+lh*8+0..7 (same as A); two column tiles per wave
    const unsigned short* bp0 = &Bs[(nw * 32 + l15) * LSTR + lh * 8];
    const unsigned short* bp1 = bp0 + 16 * LSTR;
    v8u16 b0 = *(const v8u16*)bp0;
    v8u16 b1 = *(const v8u16*)(bp0 + 16);
    v16u16 t0 = __builtin_shufflevector(b0, b1, 0,1,2,3,4,5,6,7,8,9,10,11,12,13,14,15);
    v8u16 b2 = *(const v8u16*)bp1;
    v8u16 b3 = *(const v8u16*)(bp1 + 16);
    v16u16 t1 = __builtin_shufflevector(b2, b3, 0,1,2,3,4,5,6,7,8,9,10,11,12,13,14,15);
    v16bf bf0 = __builtin_bit_cast(v16bf, t0);
    v16bf bf1 = __builtin_bit_cast(v16bf, t1);

    acc0 = __builtin_amdgcn_wmma_f32_16x16x32_bf16(false, af, false, bf0, (short)0, acc0, false, false);
    acc1 = __builtin_amdgcn_wmma_f32_16x16x32_bf16(false, af, false, bf1, (short)0, acc1, false, false);
    __syncthreads();
  }

  // C/D layout: N = lane&15, M = vgpr + 8*(lane>>4); walk rows with pointer bumps
  const int n0 = gn + nw * 32 + l15;
  const int mb = gm + mw * 16 + lh * 8;
  const float bv0 = bias ? bias[n0] : 0.f;
  const float bv1 = bias ? bias[n0 + 16] : 0.f;
  float* cp = C + (long long)mb * ldc + n0;
  const float* rp = BRES ? (Rp + (long long)mb * ldc + n0) : nullptr;
#pragma unroll
  for (int r = 0; r < 8; ++r) {
    float v0 = acc0[r] * alpha + bv0;
    float v1 = acc1[r] * alpha + bv1;
    if (BRES) { v0 += rp[0]; v1 += rp[16]; rp += ldc; }
    if (BGELU) {
      v0 = 0.5f * v0 * (1.0f + erff(v0 * 0.70710678118654752f));
      v1 = 0.5f * v1 * (1.0f + erff(v1 * 0.70710678118654752f));
    }
    cp[0] = v0;
    cp[16] = v1;
    cp += ldc;
  }
}

// ---------------- elementwise / reduction kernels ----------------
__global__ void f2bf_k(const float* __restrict__ s, unsigned short* __restrict__ d, int total) {
  int i = blockIdx.x * 256 + threadIdx.x;
  if (i < total) d[i] = f2bfs(s[i]);
}

__global__ void add_pe_k(float* __restrict__ h, int T, int D, int total) {
  int i = blockIdx.x * 256 + threadIdx.x;
  if (i >= total) return;
  int c = i % D, t = (i / D) % T;
  int i2 = c & ~1;
  float div = expf(-(float)i2 * (logf(10000.f) / (float)D));
  float ang = (float)t * div;
  h[i] += (c & 1) ? cosf(ang) : sinf(ang);
}

__global__ void query_init_k(const float* __restrict__ qe, float* __restrict__ q,
                             int T, int D, int total) {
  int i = blockIdx.x * 256 + threadIdx.x;
  if (i >= total) return;
  int c = i % D, t = (i / D) % T;
  int i2 = c & ~1;
  float div = expf(-(float)i2 * (logf(10000.f) / (float)D));
  float ang = (float)t * div;
  q[i] = qe[t * D + c] + ((c & 1) ? cosf(ang) : sinf(ang));
}

__global__ void __launch_bounds__(256)
layernorm_k(const float* __restrict__ x, const float* __restrict__ w,
            const float* __restrict__ b, float* __restrict__ y, int D) {
  const int row = blockIdx.x, tid = threadIdx.x;
  const float* xr = x + (long long)row * D;
  __shared__ float s1[256], s2[256];
  float a = 0.f, q = 0.f;
  for (int c = tid; c < D; c += 256) { float v = xr[c]; a += v; q += v * v; }
  s1[tid] = a; s2[tid] = q; __syncthreads();
  for (int s = 128; s > 0; s >>= 1) {
    if (tid < s) { s1[tid] += s1[tid + s]; s2[tid] += s2[tid + s]; }
    __syncthreads();
  }
  const float mean = s1[0] / (float)D;
  const float var  = s2[0] / (float)D - mean * mean;
  const float inv  = rsqrtf(var + 1e-5f);
  float* yr = y + (long long)row * D;
  for (int c = tid; c < D; c += 256) yr[c] = (xr[c] - mean) * inv * w[c] + b[c];
}

__global__ void __launch_bounds__(256) softmax_k(float* __restrict__ s, int Len) {
  float* r = s + (long long)blockIdx.x * Len;
  const int tid = threadIdx.x;
  __shared__ float red[256];
  float m = -3.4e38f;
  for (int c = tid; c < Len; c += 256) m = fmaxf(m, r[c]);
  red[tid] = m; __syncthreads();
  for (int st = 128; st > 0; st >>= 1) { if (tid < st) red[tid] = fmaxf(red[tid], red[tid + st]); __syncthreads(); }
  m = red[0]; __syncthreads();
  float sum = 0.f;
  for (int c = tid; c < Len; c += 256) { float e = expf(r[c] - m); r[c] = e; sum += e; }
  red[tid] = sum; __syncthreads();
  for (int st = 128; st > 0; st >>= 1) { if (tid < st) red[tid] += red[tid + st]; __syncthreads(); }
  const float inv = 1.f / red[0];
  for (int c = tid; c < Len; c += 256) r[c] *= inv;
}

__global__ void __launch_bounds__(256)
cnorm_k(const float* __restrict__ cb, float* __restrict__ cn, int E) {
  const int n = blockIdx.x, tid = threadIdx.x;
  const float* c = cb + (long long)n * E;
  __shared__ float red[256];
  float a = 0.f;
  for (int e = tid; e < E; e += 256) { float v = c[e]; a += v * v; }
  red[tid] = a; __syncthreads();
  for (int s = 128; s > 0; s >>= 1) { if (tid < s) red[tid] += red[tid + s]; __syncthreads(); }
  if (tid == 0) cn[n] = red[0];
}

__global__ void __launch_bounds__(256)
vq_k(const float* __restrict__ G, const float* __restrict__ cn,
     const float* __restrict__ ze, const float* __restrict__ cb,
     float* __restrict__ quant, int* __restrict__ idxo, float* __restrict__ acc,
     int KC, int E) {
  const int m = blockIdx.x, tid = threadIdx.x;
  const float* g = G + (long long)m * KC;
  float best = 3.4e38f; int bi = KC;
  for (int n = tid; n < KC; n += 256) {
    float d = cn[n] - 2.f * g[n];  // |c|^2 - 2 f.c
    if (d < best) { best = d; bi = n; }
  }
  __shared__ float sv[256]; __shared__ int si[256];
  sv[tid] = best; si[tid] = bi; __syncthreads();
  for (int s = 128; s > 0; s >>= 1) {
    if (tid < s) {
      if (sv[tid + s] < sv[tid] || (sv[tid + s] == sv[tid] && si[tid + s] < si[tid])) {
        sv[tid] = sv[tid + s]; si[tid] = si[tid + s];
      }
    }
    __syncthreads();
  }
  const int idx = si[0];
  if (tid == 0) idxo[m] = idx;
  const float* crow = cb + (long long)idx * E;
  float part = 0.f;
  for (int e = tid; e < E; e += 256) {
    float qv = crow[e];
    quant[(long long)m * E + e] = qv;
    float d = qv - ze[(long long)m * E + e];
    part += d * d;
  }
  __syncthreads();
  sv[tid] = part; __syncthreads();
  for (int s = 128; s > 0; s >>= 1) { if (tid < s) sv[tid] += sv[tid + s]; __syncthreads(); }
  if (tid == 0) atomicAdd(acc, sv[0]);
}

__global__ void __launch_bounds__(256)
sqdiff_k(const float* __restrict__ a, const float* __restrict__ b,
         float* __restrict__ acc, long long n) {
  __shared__ float red[256];
  const int tid = threadIdx.x;
  float v = 0.f;
  for (long long i = (long long)blockIdx.x * 256 + tid; i < n; i += (long long)gridDim.x * 256) {
    float d = a[i] - b[i]; v += d * d;
  }
  red[tid] = v; __syncthreads();
  for (int s = 128; s > 0; s >>= 1) { if (tid < s) red[tid] += red[tid + s]; __syncthreads(); }
  if (tid == 0) atomicAdd(acc, red[0]);
}

__global__ void prep_convw_k(const float* __restrict__ w, unsigned short* __restrict__ wt,
                             int E, int H, int R) {
  int i = blockIdx.x * 256 + threadIdx.x;
  if (i >= E * H * R) return;
  int e = i / (H * R), rem = i % (H * R), hh = rem / R, r = rem % R;
  wt[(long long)e * (H * R) + r * H + hh] = f2bfs(w[i]);  // w[e][hh][r] -> wt[e][(r,h)]
}

__global__ void prep_upw_k(const float* __restrict__ w, const float* __restrict__ b,
                           unsigned short* __restrict__ upt, float* __restrict__ ub,
                           int C, int O, int R) {
  int i = blockIdx.x * 256 + threadIdx.x;
  int total = C * O * R;
  if (i < total) {
    int c = i / (O * R), rem = i % (O * R), o = rem / R, k = rem % R;
    upt[((long long)(k * O + o)) * C + c] = f2bfs(w[i]);  // w[c][o][k] -> upt[(k,o)][c]
  }
  if (i < O * R) ub[i] = b[i % O];
}

__global__ void padK_k(const float* __restrict__ src, float* __restrict__ dst,
                       int Ksrc, int Kdst, int total) {
  int i = blockIdx.x * 256 + threadIdx.x;
  if (i >= total) return;
  int r = i / Kdst, c = i % Kdst;
  dst[i] = (c < Ksrc) ? src[(long long)r * Ksrc + c] : 0.f;
}

__global__ void padRows_k(const float* __restrict__ src, float* __restrict__ dst,
                          int Rsrc, int cols, int total) {
  int i = blockIdx.x * 256 + threadIdx.x;
  if (i >= total) return;
  dst[i] = ((i / cols) < Rsrc) ? src[i] : 0.f;
}

__global__ void extract_k(const float* __restrict__ src, float* __restrict__ dst,
                          int cols_src, int cols_dst, int total) {
  int i = blockIdx.x * 256 + threadIdx.x;
  if (i >= total) return;
  int r = i / cols_dst, c = i % cols_dst;
  dst[i] = src[(long long)r * cols_src + c];
}

__global__ void copy_k(const float* __restrict__ s, float* __restrict__ d, int total) {
  int i = blockIdx.x * 256 + threadIdx.x;
  if (i < total) d[i] = s[i];
}

__global__ void zero_k(float* p) { if (threadIdx.x < 2) p[threadIdx.x] = 0.f; }

__global__ void finalize_k(const float* __restrict__ acc, float* rl, float* vq,
                           float* el, float inv_valid, float inv_ne) {
  rl[0] = acc[1] * inv_valid;
  float e = acc[0] * inv_ne;
  el[0] = e;
  vq[0] = 0.25f * e;
}

// ---------------- host orchestration ----------------
extern "C" void kernel_launch(void* const* d_in, const int* in_sizes, int n_in,
                              void* d_out, int out_size, void* d_ws, size_t ws_size,
                              hipStream_t stream) {
  (void)in_sizes; (void)n_in; (void)out_size; (void)ws_size;
  auto F = [&](int i) { return (const float*)d_in[i]; };
  const float* x          = F(0);   /* mask = d_in[1] : all ones, static */
  const float* enc_in_w   = F(2);  const float* enc_in_b   = F(3);
  const float* enc_qkv_w  = F(4);  const float* enc_qkv_b  = F(5);
  const float* enc_ao_w   = F(6);  const float* enc_ao_b   = F(7);
  const float* enc_ln1_w  = F(8);  const float* enc_ln1_b  = F(9);
  const float* enc_ln2_w  = F(10); const float* enc_ln2_b  = F(11);
  const float* enc_ff1_w  = F(12); const float* enc_ff1_b  = F(13);
  const float* enc_ff2_w  = F(14); const float* enc_ff2_b  = F(15);
  const float* enc_fln_w  = F(16); const float* enc_fln_b  = F(17);
  const float* enc_conv_w = F(18); const float* enc_conv_b = F(19);
  const float* codebook   = F(20);
  const float* dec_in_w   = F(21); const float* dec_in_b   = F(22);
  const float* dec_up_w   = F(23); const float* dec_up_b   = F(24);
  const float* query_emb  = F(25);
  const float* dsa_qkv_w  = F(26); const float* dsa_qkv_b  = F(27);
  const float* dsa_o_w    = F(28); const float* dsa_o_b    = F(29);
  const float* dca_qkv_w  = F(30); const float* dca_qkv_b  = F(31);
  const float* dca_o_w    = F(32); const float* dca_o_b    = F(33);
  const float* dec_ln1_w  = F(34); const float* dec_ln1_b  = F(35);
  const float* dec_ln2_w  = F(36); const float* dec_ln2_b  = F(37);
  const float* dec_ln3_w  = F(38); const float* dec_ln3_b  = F(39);
  const float* dec_ff1_w  = F(40); const float* dec_ff1_b  = F(41);
  const float* dec_ff2_w  = F(42); const float* dec_ff2_b  = F(43);
  const float* dec_out_w  = F(44); const float* dec_out_b  = F(45);

  const int Bb = 8, T = 512, DP = 150, H = 512, Lc = 2, NH = 8, R = 4, E = 512,
            KC = 4096, FFD = 2048;
  const int DPK = 160;          // DP padded to K multiple of 32
  const int DPN = 192;          // DP padded to N multiple of 64
  const int M4 = Bb * T;        // 4096 rows
  const int M1 = Bb * T / R;    // 1024 code rows
  const int DH = H / NH;        // 64

  // f32 workspace carve-out; all sizes multiples of 4 -> every base 16B aligned
  size_t off = 0;
  auto alloc = [&](size_t n) { float* p = (float*)d_ws + off; off += n; return p; };
  float* h     = alloc((size_t)M4 * H);
  float* y     = alloc((size_t)M4 * H);
  float* qkv   = alloc((size_t)M4 * 3 * H);
  float* sc    = alloc((size_t)Bb * NH * T * T);
  float* ao    = alloc((size_t)M4 * H);
  float* ffb   = alloc((size_t)M4 * FFD);
  float* ub    = alloc((size_t)R * H);
  float* cn    = alloc(KC);
  float* G     = alloc((size_t)M1 * KC);
  float* qt    = alloc((size_t)M1 * E);
  float* mem1  = alloc((size_t)M1 * H);
  float* mem2  = alloc((size_t)M4 * H);
  float* qb    = alloc((size_t)M4 * H);
  float* cqb   = alloc((size_t)M4 * H);
  float* ckb   = alloc((size_t)M4 * H);
  float* cvb   = alloc((size_t)M4 * H);
  float* xp    = alloc((size_t)M4 * DPK);   // x padded K 150->160
  float* wp    = alloc((size_t)H * DPK);    // enc_in_w padded (f32 staging)
  float* owp   = alloc((size_t)DPN * H);    // dec_out_w padded rows 150->192 (f32 staging)
  float* obp   = alloc(DPN);                // dec_out_b padded
  float* recp  = alloc((size_t)M4 * DPN);   // padded recon
  float* zebuf = alloc((size_t)M1 * E);     // aligned z_e
  float* acc   = alloc(4);  // [0]=e_latent sum, [1]=recon sq-err sum

  // bf16 weight pool (after the f32 pool; base stays 16B aligned)
  unsigned short* hbase = (unsigned short*)((float*)d_ws + off);
  size_t hoff = 0;
  auto halloc = [&](size_t n) { unsigned short* p = hbase + hoff; hoff += (n + 7) & ~(size_t)7; return p; };
  unsigned short* wpb   = halloc((size_t)H * DPK);
  unsigned short* eqkvb = halloc((size_t)Lc * 3 * H * H);
  unsigned short* eaob  = halloc((size_t)Lc * H * H);
  unsigned short* eff1b = halloc((size_t)Lc * FFD * H);
  unsigned short* eff2b = halloc((size_t)Lc * H * FFD);
  unsigned short* wtb   = halloc((size_t)E * H * R);
  unsigned short* cbb   = halloc((size_t)KC * E);
  unsigned short* dinb  = halloc((size_t)H * E);
  unsigned short* uptb  = halloc((size_t)R * H * H);
  unsigned short* dsaqb = halloc((size_t)Lc * 3 * H * H);
  unsigned short* dsaob = halloc((size_t)Lc * H * H);
  unsigned short* dcaqb = halloc((size_t)Lc * 3 * H * H);
  unsigned short* dcaob = halloc((size_t)Lc * H * H);
  unsigned short* dff1b = halloc((size_t)Lc * FFD * H);
  unsigned short* dff2b = halloc((size_t)Lc * H * FFD);
  unsigned short* owpb  = halloc((size_t)DPN * H);

  // output layout: recon | idx(int) | recon_loss | vq_loss | e_latent | z_e
  float* out   = (float*)d_out;
  float* recon = out;
  int*   idxo  = (int*)(out + (size_t)M4 * DP);
  float* rl    = out + (size_t)M4 * DP + M1;
  float* vql   = rl + 1;
  float* ela   = rl + 2;
  float* ze    = rl + 3;

  // mode: 0 = NT f32-B, 1 = NN f32-B, 2 = NT bf16-B (TDM); gelu/res only with mode 2
  auto gemm = [&](int mode, bool gelu, bool res, const float* A, const void* Bm,
                  const float* bias, const float* resid, float* C, int M, int N, int K,
                  int lda, int ldb, int ldc, float alpha, int batch, long long aO,
                  long long aI, long long bO, long long bI, long long cO,
                  long long cI, int inner) {
    dim3 g(N / BN, M / BM, batch);
    if (mode == 0)
      gemm_wmma<0, false, false><<<g, 256, 0, stream>>>(A, Bm, bias, resid, C, M, N, K,
          lda, ldb, ldc, aO, aI, bO, bI, cO, cI, inner, alpha);
    else if (mode == 1)
      gemm_wmma<1, false, false><<<g, 256, 0, stream>>>(A, Bm, bias, resid, C, M, N, K,
          lda, ldb, ldc, aO, aI, bO, bI, cO, cI, inner, alpha);
    else if (gelu)
      gemm_wmma<2, true, false><<<g, 256, 0, stream>>>(A, Bm, bias, resid, C, M, N, K,
          lda, ldb, ldc, aO, aI, bO, bI, cO, cI, inner, alpha);
    else if (res)
      gemm_wmma<2, false, true><<<g, 256, 0, stream>>>(A, Bm, bias, resid, C, M, N, K,
          lda, ldb, ldc, aO, aI, bO, bI, cO, cI, inner, alpha);
    else
      gemm_wmma<2, false, false><<<g, 256, 0, stream>>>(A, Bm, bias, resid, C, M, N, K,
          lda, ldb, ldc, aO, aI, bO, bI, cO, cI, inner, alpha);
  };
  auto cvt = [&](const float* s, unsigned short* d, size_t n) {
    f2bf_k<<<((int)n + 255) / 256, 256, 0, stream>>>(s, d, (int)n);
  };

  const long long Tq = (long long)T * 3 * H;   // per-batch stride inside fused QKV
  const long long Th = (long long)T * H;       // per-batch stride in (B,T,H)
  const long long Ts = (long long)T * T;       // per-head score stride
  const float iscale = 0.125f;                 // 1/sqrt(64)
  const long long Z = 0;

  zero_k<<<1, 32, 0, stream>>>(acc);
  // ---- weight prep: pad + one-time f32->bf16 conversion (TDM-ready layouts) ----
  { int t = E * H * R; prep_convw_k<<<(t + 255) / 256, 256, 0, stream>>>(enc_conv_w, wtb, E, H, R); }
  { int t = H * H * R; prep_upw_k<<<(t + 255) / 256, 256, 0, stream>>>(dec_up_w, dec_up_b, uptb, ub, H, H, R); }
  cnorm_k<<<KC, 256, 0, stream>>>(codebook, cn, E);
  { int t = M4 * DPK; padK_k<<<(t + 255) / 256, 256, 0, stream>>>(x, xp, DP, DPK, t); }
  { int t = H * DPK;  padK_k<<<(t + 255) / 256, 256, 0, stream>>>(enc_in_w, wp, DP, DPK, t); }
  { int t = DPN * H;  padRows_k<<<(t + 255) / 256, 256, 0, stream>>>(dec_out_w, owp, DP, H, t); }
  padK_k<<<1, 256, 0, stream>>>(dec_out_b, obp, DP, DPN, DPN);
  cvt(wp, wpb, (size_t)H * DPK);
  cvt(enc_qkv_w, eqkvb, (size_t)Lc * 3 * H * H);
  cvt(enc_ao_w, eaob, (size_t)Lc * H * H);
  cvt(enc_ff1_w, eff1b, (size_t)Lc * FFD * H);
  cvt(enc_ff2_w, eff2b, (size_t)Lc * H * FFD);
  cvt(codebook, cbb, (size_t)KC * E);
  cvt(dec_in_w, dinb, (size_t)H * E);
  cvt(dsa_qkv_w, dsaqb, (size_t)Lc * 3 * H * H);
  cvt(dsa_o_w, dsaob, (size_t)Lc * H * H);
  cvt(dca_qkv_w, dcaqb, (size_t)Lc * 3 * H * H);
  cvt(dca_o_w, dcaob, (size_t)Lc * H * H);
  cvt(dec_ff1_w, dff1b, (size_t)Lc * FFD * H);
  cvt(dec_ff2_w, dff2b, (size_t)Lc * H * FFD);
  cvt(owp, owpb, (size_t)DPN * H);

  // ----- encoder -----
  gemm(2, false, false, xp, wpb, enc_in_b, nullptr, h, M4, H, DPK, DPK, DPK, H, 1.f,
       1, Z, Z, Z, Z, Z, Z, 1);
  add_pe_k<<<(M4 * H + 255) / 256, 256, 0, stream>>>(h, T, H, M4 * H);

  for (int l = 0; l < Lc; ++l) {
    layernorm_k<<<M4, 256, 0, stream>>>(h, enc_ln1_w + l * H, enc_ln1_b + l * H, y, H);
    gemm(2, false, false, y, eqkvb + (size_t)l * 3 * H * H, enc_qkv_b + (size_t)l * 3 * H,
         nullptr, qkv, M4, 3 * H, H, H, H, 3 * H, 1.f, 1, Z, Z, Z, Z, Z, Z, 1);
    gemm(0, false, false, qkv, qkv + H, nullptr, nullptr, sc, T, T, DH, 3 * H, 3 * H, T,
         iscale, Bb * NH, Tq, DH, Tq, DH, (long long)NH * Ts, Ts, NH);
    softmax_k<<<Bb * NH * T, 256, 0, stream>>>(sc, T);
    gemm(1, false, false, sc, qkv + 2 * H, nullptr, nullptr, ao, T, DH, T, T, 3 * H, H,
         1.f, Bb * NH, (long long)NH * Ts, Ts, Tq, DH, Th, DH, NH);
    gemm(2, false, true, ao, eaob + (size_t)l * H * H, enc_ao_b + l * H, h, h,
         M4, H, H, H, H, H, 1.f, 1, Z, Z, Z, Z, Z, Z, 1);
    layernorm_k<<<M4, 256, 0, stream>>>(h, enc_ln2_w + l * H, enc_ln2_b + l * H, y, H);
    gemm(2, true, false, y, eff1b + (size_t)l * FFD * H, enc_ff1_b + l * FFD, nullptr,
         ffb, M4, FFD, H, H, H, FFD, 1.f, 1, Z, Z, Z, Z, Z, Z, 1);
    gemm(2, false, true, ffb, eff2b + (size_t)l * H * FFD, enc_ff2_b + l * H, h, h,
         M4, H, FFD, FFD, FFD, H, 1.f, 1, Z, Z, Z, Z, Z, Z, 1);
  }
  layernorm_k<<<M4, 256, 0, stream>>>(h, enc_fln_w, enc_fln_b, y, H);
  // strided conv == GEMM on (B*T/R, R*H)
  gemm(2, false, false, y, wtb, enc_conv_b, nullptr, zebuf, M1, E, R * H, R * H, R * H,
       E, 1.f, 1, Z, Z, Z, Z, Z, Z, 1);
  copy_k<<<(M1 * E + 255) / 256, 256, 0, stream>>>(zebuf, ze, M1 * E);

  // ----- vector quantizer -----
  gemm(2, false, false, zebuf, cbb, nullptr, nullptr, G, M1, KC, E, E, E, KC, 1.f,
       1, Z, Z, Z, Z, Z, Z, 1);
  vq_k<<<M1, 256, 0, stream>>>(G, cn, zebuf, codebook, qt, idxo, acc, KC, E);

  // ----- decoder memory path -----
  gemm(2, false, false, qt, dinb, dec_in_b, nullptr, mem1, M1, H, E, E, E, H, 1.f,
       1, Z, Z, Z, Z, Z, Z, 1);
  gemm(2, false, false, mem1, uptb, ub, nullptr, mem2, M1, R * H, H, H, H, R * H, 1.f,
       1, Z, Z, Z, Z, Z, Z, 1);
  add_pe_k<<<(M4 * H + 255) / 256, 256, 0, stream>>>(mem2, T, H, M4 * H);
  query_init_k<<<(M4 * H + 255) / 256, 256, 0, stream>>>(query_emb, qb, T, H, M4 * H);

  for (int l = 0; l < Lc; ++l) {
    // self-attention
    layernorm_k<<<M4, 256, 0, stream>>>(qb, dec_ln1_w + l * H, dec_ln1_b + l * H, y, H);
    gemm(2, false, false, y, dsaqb + (size_t)l * 3 * H * H, dsa_qkv_b + (size_t)l * 3 * H,
         nullptr, qkv, M4, 3 * H, H, H, H, 3 * H, 1.f, 1, Z, Z, Z, Z, Z, Z, 1);
    gemm(0, false, false, qkv, qkv + H, nullptr, nullptr, sc, T, T, DH, 3 * H, 3 * H, T,
         iscale, Bb * NH, Tq, DH, Tq, DH, (long long)NH * Ts, Ts, NH);
    softmax_k<<<Bb * NH * T, 256, 0, stream>>>(sc, T);
    gemm(1, false, false, sc, qkv + 2 * H, nullptr, nullptr, ao, T, DH, T, T, 3 * H, H,
         1.f, Bb * NH, (long long)NH * Ts, Ts, Tq, DH, Th, DH, NH);
    gemm(2, false, true, ao, dsaob + (size_t)l * H * H, dsa_o_b + l * H, qb, qb,
         M4, H, H, H, H, H, 1.f, 1, Z, Z, Z, Z, Z, Z, 1);
    // cross-attention: Q from LN(q), K/V from mem
    const unsigned short* cwb = dcaqb + (size_t)l * 3 * H * H;
    const float* cb2 = dca_qkv_b + (size_t)l * 3 * H;
    layernorm_k<<<M4, 256, 0, stream>>>(qb, dec_ln2_w + l * H, dec_ln2_b + l * H, y, H);
    gemm(2, false, false, y,    cwb,                     cb2,         nullptr, cqb, M4, H, H, H, H, H, 1.f, 1, Z, Z, Z, Z, Z, Z, 1);
    gemm(2, false, false, mem2, cwb + (size_t)H * H,     cb2 + H,     nullptr, ckb, M4, H, H, H, H, H, 1.f, 1, Z, Z, Z, Z, Z, Z, 1);
    gemm(2, false, false, mem2, cwb + (size_t)2 * H * H, cb2 + 2 * H, nullptr, cvb, M4, H, H, H, H, H, 1.f, 1, Z, Z, Z, Z, Z, Z, 1);
    gemm(0, false, false, cqb, ckb, nullptr, nullptr, sc, T, T, DH, H, H, T, iscale,
         Bb * NH, Th, DH, Th, DH, (long long)NH * Ts, Ts, NH);
    softmax_k<<<Bb * NH * T, 256, 0, stream>>>(sc, T);
    gemm(1, false, false, sc, cvb, nullptr, nullptr, ao, T, DH, T, T, H, H, 1.f,
         Bb * NH, (long long)NH * Ts, Ts, Th, DH, Th, DH, NH);
    gemm(2, false, true, ao, dcaob + (size_t)l * H * H, dca_o_b + l * H, qb, qb,
         M4, H, H, H, H, H, 1.f, 1, Z, Z, Z, Z, Z, Z, 1);
    // FFN
    layernorm_k<<<M4, 256, 0, stream>>>(qb, dec_ln3_w + l * H, dec_ln3_b + l * H, y, H);
    gemm(2, true, false, y, dff1b + (size_t)l * FFD * H, dec_ff1_b + l * FFD, nullptr,
         ffb, M4, FFD, H, H, H, FFD, 1.f, 1, Z, Z, Z, Z, Z, Z, 1);
    gemm(2, false, true, ffb, dff2b + (size_t)l * H * FFD, dec_ff2_b + l * H, qb, qb,
         M4, H, FFD, FFD, FFD, H, 1.f, 1, Z, Z, Z, Z, Z, Z, 1);
  }

  // output projection (padded N) + extract + losses
  gemm(2, false, false, qb, owpb, obp, nullptr, recp, M4, DPN, H, H, H, DPN, 1.f,
       1, Z, Z, Z, Z, Z, Z, 1);
  { int t = M4 * DP; extract_k<<<(t + 255) / 256, 256, 0, stream>>>(recp, recon, DPN, DP, t); }
  sqdiff_k<<<1024, 256, 0, stream>>>(recon, x, acc + 1, (long long)M4 * DP);
  finalize_k<<<1, 1, 0, stream>>>(acc, rl, vql, ela, 1.f / (float)M4,
                                  1.f / ((float)M1 * (float)E));
}
